// BayesFilter_84104049590602
// MI455X (gfx1250) — matrix operand order repaired and checked by reading, not compiled, not run on hardware
//
#include <hip/hip_runtime.h>
#include <math.h>

#define B_SZ 8192
#define T_SZ 128
#define X_SZ 32
#define U_SZ 8
#define Z_SZ 3
#define W_SZ 6
#define H_SZ 128
#define M_SZ 16

typedef __attribute__((ext_vector_type(16))) _Float16 v16h;
typedef __attribute__((ext_vector_type(8)))  float    v8f;

// ---------------------------------------------------------------------------
// Activation helpers (applied on C-fragment store)
// ---------------------------------------------------------------------------
template<int ACT>
__device__ __forceinline__ float actf(float x) {
    if (ACT == 1) return fmaxf(x, 0.0f);                    // relu
    if (ACT == 2) return 1.0f / (1.0f + __expf(-x));        // sigmoid
    if (ACT == 3) return __logf(1.0f + __expf(x));          // softplus
    return x;
}

// ---------------------------------------------------------------------------
// Packed-fragment layouts (wave32 WMMA f16 16x16x32):
//  A (16xK): frag kc holds k in [kc*32, kc*32+32); lane l&15 = row;
//            half-wave (l>>4) owns a contiguous 16-wide K half; element e:
//            k = kc*32 + (l>>4)*16 + e
//            stored contiguously: pA[(kc*32 + lane)*16 + e]  -> one v16h load
//  B (KxN): frag (nt,kc): lane&15 = col; k = kc*32 + (lane>>4)*16 + e
//            stored contiguously: wt[((nt*kchunks+kc)*32 + lane)*16 + e]
//  C/D:     col = lane&15; row = r + 8*(lane>>4)
// ---------------------------------------------------------------------------

// Store f32 value for logical (row, k) into packed f16 A buffer.
__device__ __forceinline__ void packA_store(_Float16* p, int row, int k, float v) {
    int kc = k >> 5, kk = k & 31;
    int lane = row + ((kk >> 4) << 4);            // kk in [16,32) -> upper half-wave
    int e = kk & 15;
    p[((kc << 5) + lane) * 16 + e] = (_Float16)v;
}

// GEMM: C(16 x npad, f32 linear) = act( A(packed f16) * wt(packed f16) + bias )
template<int ACT>
__device__ __forceinline__ void gemm_f32(const _Float16* pA, int kpad,
                                         const _Float16* wt, int npad,
                                         const float* bias, int bias_n,
                                         float* sC, int ldc) {
    const int lane  = threadIdx.x & 31;
    const int col0  = lane & 15;
    const int rbase = (lane >> 4) << 3;
    const int kchunks = kpad >> 5;
    for (int nt = 0; nt < (npad >> 4); ++nt) {
        float bv = 0.0f;
        if (bias && (nt * 16 + col0) < bias_n) bv = bias[nt * 16 + col0];
        v8f c;
        #pragma unroll
        for (int r = 0; r < 8; ++r) c[r] = bv;
        for (int kc = 0; kc < kchunks; ++kc) {
            v16h a = *(const v16h*)(pA + (((kc << 5) + lane) << 4));
            v16h b = *(const v16h*)(wt + ((((nt * kchunks + kc) << 5) + lane) << 4));
            c = __builtin_amdgcn_wmma_f32_16x16x32_f16(
                    false, a, false, b, (short)0, c, false, false);
        }
        #pragma unroll
        for (int r = 0; r < 8; ++r)
            sC[(rbase + r) * ldc + nt * 16 + col0] = actf<ACT>(c[r]);
    }
}

// GEMM: pOut(packed f16 A-layout) = act( A(packed f16) * wt(packed f16) + bias )
template<int ACT>
__device__ __forceinline__ void gemm_pk(const _Float16* pA, int kpad,
                                        const _Float16* wt, int npad,
                                        const float* bias, int bias_n,
                                        _Float16* pOut) {
    const int lane  = threadIdx.x & 31;
    const int col0  = lane & 15;
    const int rbase = (lane >> 4) << 3;
    const int kchunks = kpad >> 5;
    for (int nt = 0; nt < (npad >> 4); ++nt) {
        float bv = 0.0f;
        if (bias && (nt * 16 + col0) < bias_n) bv = bias[nt * 16 + col0];
        v8f c;
        #pragma unroll
        for (int r = 0; r < 8; ++r) c[r] = bv;
        for (int kc = 0; kc < kchunks; ++kc) {
            v16h a = *(const v16h*)(pA + (((kc << 5) + lane) << 4));
            v16h b = *(const v16h*)(wt + ((((nt * kchunks + kc) << 5) + lane) << 4));
            c = __builtin_amdgcn_wmma_f32_16x16x32_f16(
                    false, a, false, b, (short)0, c, false, false);
        }
        #pragma unroll
        for (int r = 0; r < 8; ++r)
            packA_store(pOut, rbase + r, nt * 16 + col0, actf<ACT>(c[r]));
    }
}

// ---------------------------------------------------------------------------
// Weight packing: W (O x K, row-major f32) -> fragment-ordered f16 B operand
// ---------------------------------------------------------------------------
__global__ void pack_wt(const float* __restrict__ Wsrc, int O, int K,
                        int Kpad, int Npad, _Float16* __restrict__ dst) {
    int i = blockIdx.x * blockDim.x + threadIdx.x;
    if (i >= Kpad * Npad) return;
    int e    = i & 15;
    int lane = (i >> 4) & 31;
    int frag = i >> 9;
    int kchunks = Kpad >> 5;
    int nt = frag / kchunks, kc = frag % kchunks;
    int k = (kc << 5) + ((lane >> 4) << 4) + e;
    int n = (nt << 4) + (lane & 15);
    float v = (k < K && n < O) ? Wsrc[n * K + k] : 0.0f;
    dst[i] = (_Float16)v;
}

// Fused mixture tensor [m][ A(9) | B(24) | C(18) ] -> fragment-ordered f16 (32x64)
__global__ void pack_qabc(const float* __restrict__ qA, const float* __restrict__ qB,
                          const float* __restrict__ qC, _Float16* __restrict__ dst) {
    int i = blockIdx.x * blockDim.x + threadIdx.x;
    if (i >= 32 * 64) return;
    int e    = i & 15;
    int lane = (i >> 4) & 31;
    int nt   = i >> 9;                 // kchunks == 1
    int k = ((lane >> 4) << 4) + e;
    int n = (nt << 4) + (lane & 15);
    float v = 0.0f;
    if (k < M_SZ) {
        if (n < 9)       v = qA[k * 9  + n];
        else if (n < 33) v = qB[k * 24 + (n - 9)];
        else if (n < 51) v = qC[k * 18 + (n - 33)];
    }
    dst[i] = (_Float16)v;
}

// ---------------------------------------------------------------------------
// Bidirectional GRU over the batch axis. grid=2 (dir), block=384.
// Thread j owns gate-row j: Wh row held in VGPRs; h, gates staged in LDS.
// ---------------------------------------------------------------------------
__global__ void __launch_bounds__(384) gru_scan(
    const float* __restrict__ x,
    const float* __restrict__ Wi_f, const float* __restrict__ Wh_f,
    const float* __restrict__ bi_f, const float* __restrict__ bh_f,
    const float* __restrict__ Wi_b, const float* __restrict__ Wh_b,
    const float* __restrict__ bi_b, const float* __restrict__ bh_b,
    float* __restrict__ hs_f, float* __restrict__ hs_b)
{
    const int dir = blockIdx.x;
    const float* Wi = dir ? Wi_b : Wi_f;
    const float* Wh = dir ? Wh_b : Wh_f;
    const float* bi = dir ? bi_b : bi_f;
    const float* bh = dir ? bh_b : bh_f;
    float* hs = dir ? hs_b : hs_f;

    __shared__ float h[H_SZ];
    __shared__ float gbuf[3 * H_SZ];
    __shared__ float pbuf[H_SZ];
    __shared__ float xloc[X_SZ];

    const int j = threadIdx.x;   // 0..383
    float rw[H_SZ];
    #pragma unroll
    for (int k = 0; k < H_SZ; ++k) rw[k] = Wh[j * H_SZ + k];
    float wir[X_SZ];
    #pragma unroll
    for (int k = 0; k < X_SZ; ++k) wir[k] = Wi[j * X_SZ + k];
    const float bij = bi[j];
    const float bhj = bh[j];
    if (j < H_SZ) h[j] = 0.0f;
    __syncthreads();

    for (int s = 0; s < B_SZ; ++s) {
        const int b = dir ? (B_SZ - 1 - s) : s;
        if (j < X_SZ) xloc[j] = x[(size_t)b * T_SZ * X_SZ + j];   // x[b,0,:]
        __syncthreads();
        float pre = bij;
        #pragma unroll
        for (int k = 0; k < X_SZ; ++k) pre += wir[k] * xloc[k];
        float gh = bhj;
        #pragma unroll 16
        for (int k = 0; k < H_SZ; ++k) gh += rw[k] * h[k];
        if (j < 2 * H_SZ) { gbuf[j] = pre + gh; }
        else              { gbuf[j] = gh; pbuf[j - 2 * H_SZ] = pre; }
        __syncthreads();
        if (j < H_SZ) {
            float r  = 1.0f / (1.0f + __expf(-gbuf[j]));
            float zg = 1.0f / (1.0f + __expf(-gbuf[H_SZ + j]));
            float n  = tanhf(pbuf[j] + r * gbuf[2 * H_SZ + j]);
            float hn = (1.0f - zg) * n + zg * h[j];
            h[j] = hn;
            hs[(size_t)b * H_SZ + j] = hn;
        }
        __syncthreads();
    }
}

// ---------------------------------------------------------------------------
// Encoder head: per 16-batch tile (1 wave). Produces t=0 outputs and z0.
// ---------------------------------------------------------------------------
__global__ void __launch_bounds__(32) encoder_head(
    const float* __restrict__ hs_f, const float* __restrict__ hs_b,
    const _Float16* wt_pxi, const float* pxi_b,
    const _Float16* wt_mu,  const float* mu_b,
    const _Float16* wt_sg,  const float* sg_b,
    const float* __restrict__ eps_w1,
    const _Float16* wt_pl1, const float* pl_b1,
    const _Float16* wt_pl2, const float* pl_b2,
    const _Float16* wt_pth1, const float* pth_b1,
    const _Float16* wt_pth2, const float* pth_b2,
    float* __restrict__ z0ws, float* __restrict__ xp,
    float* __restrict__ wd, float* __restrict__ zp)
{
    __shared__ __attribute__((aligned(32))) _Float16 pA[256 * 16];  // packed A (Kpad<=256)
    __shared__ __attribute__((aligned(32))) _Float16 pH[128 * 16];  // packed hidden
    __shared__ float sGM[16 * 16];
    __shared__ float sGS[16 * 16];
    __shared__ float sXP[16 * 32];
    const int lane = threadIdx.x;
    const int b0 = blockIdx.x * 16;

    for (int i = lane; i < 16 * 256; i += 32) {                     // bi_out, packed
        int r = i >> 8, c = i & 255;
        float v = (c < H_SZ) ? hs_f[(size_t)(b0 + r) * H_SZ + c]
                             : hs_b[(size_t)(b0 + r) * H_SZ + (c - H_SZ)];
        packA_store(pA, r, c, v);
    }
    __syncthreads();
    gemm_pk<1>(pA, 256, wt_pxi, 128, pxi_b, 128, pH);               // h = relu(...)
    __syncthreads();
    gemm_f32<0>(pH, 128, wt_mu, 16, mu_b, W_SZ, sGM, 16);           // mu1
    gemm_f32<3>(pH, 128, wt_sg, 16, sg_b, W_SZ, sGS, 16);           // sg1 = softplus
    __syncthreads();
    for (int i = lane; i < 16 * 32; i += 32) {                      // w1 + t=0 w_dists
        int r = i >> 5, c = i & 31;
        float v = 0.0f;
        if (c < W_SZ) {
            float m = sGM[r * 16 + c], s = sGS[r * 16 + c];
            v = m + s * eps_w1[(size_t)(b0 + r) * W_SZ + c];
            size_t base = (size_t)(b0 + r) * T_SZ * W_SZ * 2 + (size_t)c * 2;
            wd[base] = m; wd[base + 1] = s;
        }
        packA_store(pA, r, c, v);
    }
    __syncthreads();
    gemm_pk<0>(pA, 32, wt_pl1, 128, pl_b1, 128, pH);
    __syncthreads();
    gemm_f32<0>(pH, 128, wt_pl2, 16, pl_b2, Z_SZ, sGM, 16);         // z0
    __syncthreads();
    for (int i = lane; i < 16 * 32; i += 32) {
        int r = i >> 5, c = i & 31;
        float v = 0.0f;
        if (c < Z_SZ) {
            v = sGM[r * 16 + c];
            z0ws[(size_t)(b0 + r) * Z_SZ + c] = v;
            zp[(size_t)(b0 + r) * T_SZ * Z_SZ + c] = v;
        }
        packA_store(pA, r, c, v);
    }
    __syncthreads();
    gemm_pk<0>(pA, 32, wt_pth1, 128, pth_b1, 128, pH);
    __syncthreads();
    gemm_f32<0>(pH, 128, wt_pth2, 32, pth_b2, X_SZ, sXP, 32);       // xp0
    __syncthreads();
    for (int i = lane; i < 16 * 32; i += 32) {
        int r = i >> 5, c = i & 31;
        xp[(size_t)(b0 + r) * T_SZ * X_SZ + c] = sXP[i];
    }
}

// ---------------------------------------------------------------------------
// Time recurrence: per 16-batch tile (1 wave) runs t = 1..T-1 sequentially.
// ---------------------------------------------------------------------------
__global__ void __launch_bounds__(32) scan_steps(
    const float* __restrict__ x, const float* __restrict__ u,
    const float* __restrict__ eps_w, const float* __restrict__ z0ws,
    const _Float16* wt_f1, const float* fpsi_b1,
    const _Float16* wt_f2, const float* fpsi_b2,
    const _Float16* wt_qabc,
    const _Float16* wt_m1, const float* qm_b1,
    const _Float16* wt_m2, const float* qm_b2,
    const _Float16* wt_s1, const float* qs_b1,
    const _Float16* wt_s2, const float* qs_b2,
    const _Float16* wt_pth1, const float* pth_b1,
    const _Float16* wt_pth2, const float* pth_b2,
    float* __restrict__ xp, float* __restrict__ wd, float* __restrict__ zp)
{
    __shared__ __attribute__((aligned(32))) _Float16 pX[64 * 16];   // packed A inputs
    __shared__ __attribute__((aligned(32))) _Float16 pH[128 * 16];  // packed hidden
    __shared__ __attribute__((aligned(32))) _Float16 pS[32 * 16];   // packed alpha
    __shared__ float sS[16 * 16];       // logits (f32)
    __shared__ float sABC[16 * 64];
    __shared__ float sGM[16 * 16];
    __shared__ float sGS[16 * 16];
    __shared__ float sZ[16 * Z_SZ];
    __shared__ float sXP[16 * 32];
    const int lane = threadIdx.x;
    const int b0 = blockIdx.x * 16;

    for (int i = lane; i < 32 * 16; i += 32) pS[i] = (_Float16)0.0f;  // K-pad for alpha
    for (int i = lane; i < 16 * Z_SZ; i += 32) sZ[i] = z0ws[(size_t)b0 * Z_SZ + i];
    __syncthreads();

    for (int t = 1; t < T_SZ; ++t) {
        // --- alpha = softmax(sigmoid([z,u] W1^T + b1) W2^T + b2) ---
        for (int i = lane; i < 16 * 32; i += 32) {
            int r = i >> 5, c = i & 31;
            float v = 0.0f;
            if (c < Z_SZ) v = sZ[r * Z_SZ + c];
            else if (c < Z_SZ + U_SZ)
                v = u[(size_t)(b0 + r) * T_SZ * U_SZ + (size_t)(t - 1) * U_SZ + (c - Z_SZ)];
            packA_store(pX, r, c, v);
        }
        __syncthreads();
        gemm_pk<2>(pX, 32, wt_f1, 128, fpsi_b1, 128, pH);
        __syncthreads();
        gemm_f32<0>(pH, 128, wt_f2, 16, fpsi_b2, M_SZ, sS, 16);
        __syncthreads();
        if (lane < 16) {
            float mx = -1e30f;
            for (int c = 0; c < 16; ++c) mx = fmaxf(mx, sS[lane * 16 + c]);
            float sum = 0.0f;
            float ex[16];
            for (int c = 0; c < 16; ++c) { ex[c] = __expf(sS[lane * 16 + c] - mx); sum += ex[c]; }
            float inv = 1.0f / sum;
            for (int c = 0; c < 16; ++c) packA_store(pS, lane, c, ex[c] * inv);
        }
        __syncthreads();
        // --- [A|B|C] = alpha @ q[A|B|C] (fused, no bias) ---
        gemm_f32<0>(pS, 32, wt_qabc, 64, (const float*)nullptr, 0, sABC, 64);
        // --- zxu = [z, x_t, u_{t-1}] (packed, Kpad=64) ---
        for (int i = lane; i < 16 * 64; i += 32) {
            int r = i >> 6, c = i & 63;
            float v = 0.0f;
            if (c < 3) v = sZ[r * 3 + c];
            else if (c < 35) v = x[(size_t)(b0 + r) * T_SZ * X_SZ + (size_t)t * X_SZ + (c - 3)];
            else if (c < 43) v = u[(size_t)(b0 + r) * T_SZ * U_SZ + (size_t)(t - 1) * U_SZ + (c - 35)];
            packA_store(pX, r, c, v);
        }
        __syncthreads();
        gemm_pk<2>(pX, 64, wt_m1, 128, qm_b1, 128, pH);
        __syncthreads();
        gemm_f32<0>(pH, 128, wt_m2, 16, qm_b2, W_SZ, sGM, 16);   // wmu
        __syncthreads();
        gemm_pk<2>(pX, 64, wt_s1, 128, qs_b1, 128, pH);
        __syncthreads();
        gemm_f32<3>(pH, 128, wt_s2, 16, qs_b2, W_SZ, sGS, 16);   // wsg
        __syncthreads();
        // --- z' = A z + B u + C w ; emit w_dists, z_pred ---
        if (lane < 16) {
            const int l = lane;
            float w[W_SZ];
            for (int j = 0; j < W_SZ; ++j) {
                float m = sGM[l * 16 + j], s = sGS[l * 16 + j];
                float e = eps_w[((size_t)(t - 1) * B_SZ + b0 + l) * W_SZ + j];
                w[j] = m + s * e;
                size_t base = (size_t)(b0 + l) * T_SZ * W_SZ * 2 + (size_t)t * W_SZ * 2 + (size_t)j * 2;
                wd[base] = m; wd[base + 1] = s;
            }
            float uu[U_SZ];
            for (int j = 0; j < U_SZ; ++j)
                uu[j] = u[(size_t)(b0 + l) * T_SZ * U_SZ + (size_t)(t - 1) * U_SZ + j];
            float zo[Z_SZ];
            for (int j = 0; j < Z_SZ; ++j) zo[j] = sZ[l * Z_SZ + j];
            float zn[Z_SZ];
            for (int i2 = 0; i2 < Z_SZ; ++i2) {
                float acc = 0.0f;
                for (int j = 0; j < 3; ++j) acc += sABC[l * 64 + i2 * 3 + j] * zo[j];
                for (int j = 0; j < 8; ++j) acc += sABC[l * 64 + 9 + i2 * 8 + j] * uu[j];
                for (int j = 0; j < 6; ++j) acc += sABC[l * 64 + 33 + i2 * 6 + j] * w[j];
                zn[i2] = acc;
                zp[(size_t)(b0 + l) * T_SZ * Z_SZ + (size_t)t * Z_SZ + i2] = acc;
            }
            for (int i2 = 0; i2 < Z_SZ; ++i2) sZ[l * Z_SZ + i2] = zn[i2];
        }
        __syncthreads();
        // --- x_pred_t = (z' W1^T + b1) W2^T + b2 ---
        for (int i = lane; i < 16 * 32; i += 32) {
            int r = i >> 5, c = i & 31;
            packA_store(pX, r, c, (c < Z_SZ) ? sZ[r * Z_SZ + c] : 0.0f);
        }
        __syncthreads();
        gemm_pk<0>(pX, 32, wt_pth1, 128, pth_b1, 128, pH);
        __syncthreads();
        gemm_f32<0>(pH, 128, wt_pth2, 32, pth_b2, X_SZ, sXP, 32);
        __syncthreads();
        for (int i = lane; i < 16 * 32; i += 32) {
            int r = i >> 5, c = i & 31;
            xp[(size_t)(b0 + r) * T_SZ * X_SZ + (size_t)t * X_SZ + c] = sXP[i];
        }
        __syncthreads();
    }
}

// ---------------------------------------------------------------------------
extern "C" void kernel_launch(void* const* d_in, const int* in_sizes, int n_in,
                              void* d_out, int out_size, void* d_ws, size_t ws_size,
                              hipStream_t stream) {
    const float* x      = (const float*)d_in[0];
    const float* u      = (const float*)d_in[1];
    const float* eps_w1 = (const float*)d_in[2];
    const float* eps_w  = (const float*)d_in[3];
    const float* Wi_f   = (const float*)d_in[4];
    const float* Wh_f   = (const float*)d_in[5];
    const float* bi_f   = (const float*)d_in[6];
    const float* bh_f   = (const float*)d_in[7];
    const float* Wi_b   = (const float*)d_in[8];
    const float* Wh_b   = (const float*)d_in[9];
    const float* bi_b   = (const float*)d_in[10];
    const float* bh_b   = (const float*)d_in[11];
    const float* pxi_W  = (const float*)d_in[12];
    const float* pxi_b  = (const float*)d_in[13];
    const float* mu_W   = (const float*)d_in[14];
    const float* mu_b   = (const float*)d_in[15];
    const float* sg_W   = (const float*)d_in[16];
    const float* sg_b   = (const float*)d_in[17];
    const float* pl_W1  = (const float*)d_in[18];
    const float* pl_b1  = (const float*)d_in[19];
    const float* pl_W2  = (const float*)d_in[20];
    const float* pl_b2  = (const float*)d_in[21];
    const float* qA     = (const float*)d_in[22];
    const float* qB     = (const float*)d_in[23];
    const float* qC     = (const float*)d_in[24];
    const float* fpsi_W1= (const float*)d_in[25];
    const float* fpsi_b1= (const float*)d_in[26];
    const float* fpsi_W2= (const float*)d_in[27];
    const float* fpsi_b2= (const float*)d_in[28];
    const float* qm_W1  = (const float*)d_in[29];
    const float* qm_b1  = (const float*)d_in[30];
    const float* qm_W2  = (const float*)d_in[31];
    const float* qm_b2  = (const float*)d_in[32];
    const float* qs_W1  = (const float*)d_in[33];
    const float* qs_b1  = (const float*)d_in[34];
    const float* qs_W2  = (const float*)d_in[35];
    const float* qs_b2  = (const float*)d_in[36];
    const float* pth_W1 = (const float*)d_in[37];
    const float* pth_b1 = (const float*)d_in[38];
    const float* pth_W2 = (const float*)d_in[39];
    const float* pth_b2 = (const float*)d_in[40];

    float* out = (float*)d_out;
    float* xp = out;                                        // (B,T,X)
    float* wd = xp + (size_t)B_SZ * T_SZ * X_SZ;            // (B,T,W,2)
    float* zp = wd + (size_t)B_SZ * T_SZ * W_SZ * 2;        // (B,T,Z)

    char* wsb = (char*)d_ws;
    size_t off = 0;
    auto carve = [&](size_t bytes) {
        void* p = wsb + off;
        off = (off + bytes + 255) & ~(size_t)255;
        return p;
    };
    float* hs_f = (float*)carve(sizeof(float) * B_SZ * H_SZ);
    float* hs_b = (float*)carve(sizeof(float) * B_SZ * H_SZ);
    float* z0ws = (float*)carve(sizeof(float) * B_SZ * Z_SZ);
    _Float16* wt_pxi  = (_Float16*)carve(2 * 256 * 128);
    _Float16* wt_mu   = (_Float16*)carve(2 * 128 * 16);
    _Float16* wt_sg   = (_Float16*)carve(2 * 128 * 16);
    _Float16* wt_pl1  = (_Float16*)carve(2 * 32 * 128);
    _Float16* wt_pl2  = (_Float16*)carve(2 * 128 * 16);
    _Float16* wt_f1   = (_Float16*)carve(2 * 32 * 128);
    _Float16* wt_f2   = (_Float16*)carve(2 * 128 * 16);
    _Float16* wt_m1   = (_Float16*)carve(2 * 64 * 128);
    _Float16* wt_m2   = (_Float16*)carve(2 * 128 * 16);
    _Float16* wt_s1   = (_Float16*)carve(2 * 64 * 128);
    _Float16* wt_s2   = (_Float16*)carve(2 * 128 * 16);
    _Float16* wt_pth1 = (_Float16*)carve(2 * 32 * 128);
    _Float16* wt_pth2 = (_Float16*)carve(2 * 128 * 32);
    _Float16* wt_qabc = (_Float16*)carve(2 * 32 * 64);

    auto pk = [&](const float* Wm, int O, int K, int Kpad, int Npad, _Float16* dst) {
        int total = Kpad * Npad;
        pack_wt<<<(total + 255) / 256, 256, 0, stream>>>(Wm, O, K, Kpad, Npad, dst);
    };
    pk(pxi_W,  128, 256, 256, 128, wt_pxi);
    pk(mu_W,     6, 128, 128,  16, wt_mu);
    pk(sg_W,     6, 128, 128,  16, wt_sg);
    pk(pl_W1,  128,   6,  32, 128, wt_pl1);
    pk(pl_W2,    3, 128, 128,  16, wt_pl2);
    pk(fpsi_W1,128,  11,  32, 128, wt_f1);
    pk(fpsi_W2, 16, 128, 128,  16, wt_f2);
    pk(qm_W1,  128,  43,  64, 128, wt_m1);
    pk(qm_W2,    6, 128, 128,  16, wt_m2);
    pk(qs_W1,  128,  43,  64, 128, wt_s1);
    pk(qs_W2,    6, 128, 128,  16, wt_s2);
    pk(pth_W1, 128,   3,  32, 128, wt_pth1);
    pk(pth_W2,  32, 128, 128,  32, wt_pth2);
    pack_qabc<<<(32 * 64 + 255) / 256, 256, 0, stream>>>(qA, qB, qC, wt_qabc);

    gru_scan<<<2, 384, 0, stream>>>(x, Wi_f, Wh_f, bi_f, bh_f,
                                    Wi_b, Wh_b, bi_b, bh_b, hs_f, hs_b);

    encoder_head<<<B_SZ / 16, 32, 0, stream>>>(
        hs_f, hs_b, wt_pxi, pxi_b, wt_mu, mu_b, wt_sg, sg_b, eps_w1,
        wt_pl1, pl_b1, wt_pl2, pl_b2, wt_pth1, pth_b1, wt_pth2, pth_b2,
        z0ws, xp, wd, zp);

    scan_steps<<<B_SZ / 16, 32, 0, stream>>>(
        x, u, eps_w, z0ws,
        wt_f1, fpsi_b1, wt_f2, fpsi_b2, wt_qabc,
        wt_m1, qm_b1, wt_m2, qm_b2, wt_s1, qs_b1, wt_s2, qs_b2,
        wt_pth1, pth_b1, wt_pth2, pth_b2, xp, wd, zp);

    (void)in_sizes; (void)n_in; (void)out_size; (void)ws_size;
}
